// Model_4896262718009
// MI455X (gfx1250) — matrix operand order repaired and checked by reading, not compiled, not run on hardware
//
#include <hip/hip_runtime.h>
#include <hip/hip_bf16.h>
#include <cstddef>

// ---------------- types ----------------
typedef __attribute__((ext_vector_type(16))) _Float16 v16h;
typedef __attribute__((ext_vector_type(8)))  _Float16 v8h;
typedef __attribute__((ext_vector_type(8)))  float    v8f;
typedef unsigned int u32;
typedef __attribute__((ext_vector_type(4))) u32 v4u;
typedef __attribute__((ext_vector_type(8))) u32 v8u;
typedef __attribute__((ext_vector_type(4))) int i32x4;

#define NTOK   8192      // B*L
#define SEQL   1024
#define FLAG_RELU     1
#define FLAG_SOFTPLUS 2

// ---------------- CDNA5 feature detection ----------------
#ifdef __has_builtin
# if __has_builtin(__builtin_amdgcn_global_load_async_to_lds_b128)
#  define HAVE_ASYNC 1
# endif
# if __has_builtin(__builtin_amdgcn_s_wait_asynccnt)
#  define HAVE_WAIT_ASYNC 1
# endif
# if __has_builtin(__builtin_amdgcn_s_wait_tensorcnt)
#  define HAVE_WAIT_TENSOR 1
# endif
#endif
#ifndef HAVE_ASYNC
# define HAVE_ASYNC 0
#endif
#ifndef HAVE_WAIT_ASYNC
# define HAVE_WAIT_ASYNC 0
#endif
#ifndef HAVE_WAIT_TENSOR
# define HAVE_WAIT_TENSOR 0
#endif
#define USE_TDM 1   // tensor_load_to_lds (inline asm) for the uniform weight tile

typedef __attribute__((address_space(1))) i32x4 g_i32x4;
typedef __attribute__((address_space(3))) i32x4 l_i32x4;

// 16B global -> LDS copy (async on CDNA5, sync fallback)
__device__ __forceinline__ void copy16_g2l(const _Float16* g, _Float16* l) {
#if HAVE_ASYNC
  __builtin_amdgcn_global_load_async_to_lds_b128((g_i32x4*)g, (l_i32x4*)l, 0, 0);
#else
  *(v8h*)l = *(const v8h*)g;
#endif
}

__device__ __forceinline__ void wait_stage_counters() {
#if HAVE_ASYNC
# if HAVE_WAIT_ASYNC
  __builtin_amdgcn_s_wait_asynccnt(0);
# else
  asm volatile("s_wait_asynccnt 0" ::: "memory");
# endif
#endif
#if USE_TDM
# if HAVE_WAIT_TENSOR
  __builtin_amdgcn_s_wait_tensorcnt(0);
# else
  asm volatile("s_wait_tensorcnt 0" ::: "memory");
# endif
#endif
}

#if USE_TDM
// TDM: load a 64-row x 32-element f16 tile (row stride K elems) into LDS with
// 80-byte padded rows (pad_interval=16 DWORDs, pad_amount=4 DWORDs).
__device__ __forceinline__ void tdm_load_tile_a(const _Float16* gsrc, u32 lds_off, int K) {
  unsigned long long ga = (unsigned long long)(size_t)gsrc;
  v4u g0;
  g0[0] = 1u;                                            // count=1, user mode
  g0[1] = lds_off;                                       // lds_addr (bytes)
  g0[2] = (u32)ga;                                       // global_addr[31:0]
  g0[3] = ((u32)(ga >> 32) & 0x01FFFFFFu) | (2u << 30);  // addr[56:32] | type=2
  v8u g1;
  g1[0] = (1u << 16)        // data_size = 2 bytes
        | (1u << 20)        // pad_enable
        | (3u << 22)        // pad_interval = 16 DWORDs (one 64B row)
        | (3u << 25);       // pad_amount   = 4 DWORDs (16B) -> 80B row stride
  g1[1] = ((u32)K) << 16;                 // tensor_dim0[15:0]
  g1[2] = (((u32)K) >> 16) | (64u << 16); // tensor_dim0[31:16] | tensor_dim1=64
  g1[3] = (32u << 16);                    // tile_dim0 = 32
  g1[4] = 64u;                            // tile_dim1 = 64, tile_dim2 = 0
  g1[5] = (u32)K;                         // tensor_dim0_stride[31:0]
  g1[6] = 0u;
  g1[7] = 0u;
  asm volatile("tensor_load_to_lds %0, %1" :: "s"(g0), "s"(g1) : "memory");
}
typedef __attribute__((address_space(3))) _Float16 lds_f16_t;
#endif

// ============================================================
// Generic WMMA GEMM:
//   out[n][m] = act( sum_tap sum_k W[tap][m][k] * X[n - tap*dil][k] + bias[m] + add[n][m] )
// X: [N][in_stride] f16 token-major.  W: [taps][M][K] f16.
// Block tile 64(M) x 128(N), 8 waves (2x4), K step 32, double-buffered LDS.
// A tile staged by TDM (wave 0); B tile staged by per-lane async-to-LDS.
// ============================================================
__global__ __launch_bounds__(256) void wmma_gemm_kernel(
    const _Float16* __restrict__ W,
    const _Float16* __restrict__ X,
    const float*    __restrict__ bias,
    const _Float16* __restrict__ add16,
    const float*    __restrict__ addf,
    const _Float16* __restrict__ zpad,   // >=16 f16 of zeros (causal fill)
    _Float16*       __restrict__ out16,
    float*          __restrict__ outf,
    int M, int K, int in_stride, int out_stride,
    int taps, int dil, int flags)
{
  __shared__ _Float16 lA[2][64 * 40];
  __shared__ _Float16 lB[2][128 * 40];

  const int tid  = threadIdx.x;
  const int n0   = blockIdx.x * 128;
  const int m0   = blockIdx.y * 64;
  const int wave = tid >> 5;
  const int lane = tid & 31;
  const int wm   = wave & 1;
  const int wn   = wave >> 1;

  // fold m0 into the weight pointer: tile row stride is K
  const _Float16* Wt = W + (size_t)m0 * K;

  // per-thread B-loader constants (128 cols x 32 k, 32B per thread)
  const int bcol  = tid >> 1;
  const int bhalf = tid & 1;
  const int btok  = n0 + bcol;
  const int bl    = btok & (SEQL - 1);
  const _Float16* bp = X + (size_t)btok * in_stride + bhalf * 16;
  const int bofs  = bcol * 40 + bhalf * 16;

#if !USE_TDM
  const int arow = tid >> 2, apart = tid & 3;
#endif

  v8f c[2][2] = {};

  const int kiters = K >> 5;
  const int stages = taps * kiters;

  // ---- stage issue (tap, kk) into buffer buf ----
  auto issue = [&](int buf, int tap, int kk) {
#if USE_TDM
    if (wave == 0)
      tdm_load_tile_a(Wt + (size_t)tap * M * K + kk,
                      (u32)(size_t)(lds_f16_t*)&lA[buf][0], K);
#else
    {
      const v8h* src = (const v8h*)(Wt + (size_t)tap * M * K + (size_t)arow * K + kk + apart * 8);
      *(v8h*)(&lA[buf][arow * 40 + apart * 8]) = *src;
    }
#endif
    const int shift = tap * dil;
    const _Float16* src = (bl >= shift) ? (bp + kk - (size_t)shift * in_stride) : zpad;
    _Float16* dst = &lB[buf][bofs];
    copy16_g2l(src, dst);
    copy16_g2l(src + 8, dst + 8);
  };

  issue(0, 0, 0);
  int ntap = 0, nkk = 32;               // next stage to issue
  if (nkk == K) { nkk = 0; ntap = 1; }
  wait_stage_counters();
  __syncthreads();

  const int khalf8  = (lane >> 4) << 3;
  const int khalf16 = (lane >> 4) << 4;
  const int arow0 = wm * 32 + (lane & 15);
  const int bcol0 = wn * 32 + (lane & 15);

  for (int s = 0; s < stages; ++s) {
    const int cur = s & 1;
    if (s + 1 < stages) {
      issue(cur ^ 1, ntap, nkk);
      nkk += 32;
      if (nkk == K) { nkk = 0; ++ntap; }
    }
    const _Float16* A  = lA[cur];
    const _Float16* Bm = lB[cur];

    v16h a[2], b[2];
#pragma unroll
    for (int mi = 0; mi < 2; ++mi) {
      const _Float16* p = A + (arow0 + mi * 16) * 40 + khalf8;
      v8h lo = *(const v8h*)(p);
      v8h hi = *(const v8h*)(p + 16);
#pragma unroll
      for (int i = 0; i < 8; ++i) { a[mi][i] = lo[i]; a[mi][8 + i] = hi[i]; }
    }
#pragma unroll
    for (int ni = 0; ni < 2; ++ni) {
      const _Float16* p = Bm + (bcol0 + ni * 16) * 40 + khalf8;
      v8h lo = *(const v8h*)(p);
      v8h hi = *(const v8h*)(p + 16);
#pragma unroll
      for (int i = 0; i < 8; ++i) { b[ni][i] = lo[i]; b[ni][8 + i] = hi[i]; }
    }
#pragma unroll
    for (int mi = 0; mi < 2; ++mi)
#pragma unroll
      for (int ni = 0; ni < 2; ++ni)
        c[mi][ni] = __builtin_amdgcn_wmma_f32_16x16x32_f16(
            false, a[mi], false, b[ni], (short)0, c[mi][ni], false, false);
    wait_stage_counters();
    __syncthreads();
  }

  // ---- fused epilogue ----
#pragma unroll
  for (int mi = 0; mi < 2; ++mi) {
#pragma unroll
    for (int ni = 0; ni < 2; ++ni) {
      int col   = n0 + wn * 32 + ni * 16 + (lane & 15);
      int mbase = m0 + wm * 32 + mi * 16 + ((lane >> 4) << 3);
      float vals[8];
#pragma unroll
      for (int v = 0; v < 8; ++v) {
        int row = mbase + v;
        float x = c[mi][ni][v];
        if (bias) x += bias[row];
        size_t eo = (size_t)col * out_stride + row;
        if (add16) x += (float)add16[eo];
        if (addf)  x += addf[eo];
        if (flags & FLAG_RELU)     x = fmaxf(x, 0.f);
        if (flags & FLAG_SOFTPLUS) x = (x > 20.f) ? x : log1pf(__expf(x));
        vals[v] = x;
      }
      size_t base = (size_t)col * out_stride + mbase;
      if (out16) {
        v8h o;
#pragma unroll
        for (int v = 0; v < 8; ++v) o[v] = (_Float16)vals[v];
        *(v8h*)(out16 + base) = o;
      }
      if (outf) {
        v8f o;
#pragma unroll
        for (int v = 0; v < 8; ++v) o[v] = vals[v];
        *(v8f*)(outf + base) = o;
      }
    }
  }
}

// ============================================================
// small helper kernels
// ============================================================
__global__ void cast_f32_to_f16(const float* __restrict__ s, _Float16* __restrict__ d, int n) {
  int i = blockIdx.x * blockDim.x + threadIdx.x;
  if (i < n) d[i] = (_Float16)s[i];
}
__global__ void cast_f16_to_f32(const _Float16* __restrict__ s, float* __restrict__ d, int n) {
  int i = blockIdx.x * blockDim.x + threadIdx.x;
  if (i < n) d[i] = (float)s[i];
}
__global__ void add_relu_f16(const _Float16* a, const _Float16* b, _Float16* o, int n) {
  int i = blockIdx.x * blockDim.x + threadIdx.x;
  if (i < n) o[i] = (_Float16)fmaxf((float)a[i] + (float)b[i], 0.f);
}
__global__ void zero_pad_kernel(_Float16* z) { z[threadIdx.x] = (_Float16)0.f; }
// conv weight (O,I,3) f32 -> [tap][o][i] f16
__global__ void convw_to_f16(const float* __restrict__ s, _Float16* __restrict__ d, int O, int I) {
  int idx = blockIdx.x * blockDim.x + threadIdx.x;
  int tot = O * I * 3;
  if (idx >= tot) return;
  int t = idx / (O * I), r = idx % (O * I);
  int o = r / I, i = r % I;
  d[idx] = (_Float16)s[((size_t)o * I + i) * 3 + t];
}
__global__ void mat_pad_f16(const float* __restrict__ s, _Float16* __restrict__ d,
                            int M, int K, int Mp, int Kp) {
  int idx = blockIdx.x * blockDim.x + threadIdx.x;
  if (idx >= Mp * Kp) return;
  int m = idx / Kp, k = idx % Kp;
  d[idx] = (m < M && k < K) ? (_Float16)s[(size_t)m * K + k] : (_Float16)0.f;
}
__global__ void a_prep(const float* __restrict__ alog, float* __restrict__ A, int n) {
  int i = blockIdx.x * blockDim.x + threadIdx.x;
  if (i < n) A[i] = -__expf(alog[i]);
}
// depthwise causal conv (d_conv=4) + SiLU. xz: [N][1024] (xb in cols 0..511)
__global__ void dwconv_silu(const _Float16* __restrict__ xz,
                            const float* __restrict__ cw, const float* __restrict__ cb,
                            _Float16* __restrict__ xc, int N) {
  int idx = blockIdx.x * blockDim.x + threadIdx.x;
  if (idx >= N * 512) return;
  int n = idx >> 9, d = idx & 511;
  int l = n & (SEQL - 1);
  float v = cb[d];
#pragma unroll
  for (int j = 0; j < 4; ++j) {
    int li = l - 3 + j;
    if (li >= 0) v += cw[d * 4 + j] * (float)xz[(size_t)(n - 3 + j) * 1024 + d];
  }
  v = v / (1.f + __expf(-v));
  xc[(size_t)n * 512 + d] = (_Float16)v;
}
// selective scan
__global__ __launch_bounds__(512) void scan_kernel(
    const _Float16* __restrict__ xdbl, const _Float16* __restrict__ dt16,
    const _Float16* __restrict__ xc16, const _Float16* __restrict__ xz,
    const float* __restrict__ Af, const float* __restrict__ Dp,
    _Float16* __restrict__ y16)
{
  int b = blockIdx.x, d = threadIdx.x;
  __shared__ float Bs[16], Cs[16];
  float h[16];
#pragma unroll
  for (int s = 0; s < 16; ++s) h[s] = 0.f;
  float Ar[16];
#pragma unroll
  for (int s = 0; s < 16; ++s) Ar[s] = Af[d * 16 + s];
  float Dd = Dp[d];
  for (int l = 0; l < SEQL; ++l) {
    int n = b * SEQL + l;
    if (d < 32) {
      int s = d & 15;
      float v = (float)xdbl[(size_t)n * 64 + 16 + ((d < 16) ? 0 : 16) + s];
      if (d < 16) Bs[s] = v; else Cs[s] = v;
    }
    __syncthreads();
    float dt = (float)dt16[(size_t)n * 512 + d];
    float x  = (float)xc16[(size_t)n * 512 + d];
    float y = 0.f;
#pragma unroll
    for (int s = 0; s < 16; ++s) {
      h[s] = __expf(dt * Ar[s]) * h[s] + dt * Bs[s] * x;
      y += h[s] * Cs[s];
    }
    y += x * Dd;
    float z = (float)xz[(size_t)n * 1024 + 512 + d];
    y *= z / (1.f + __expf(-z));
    y16[(size_t)n * 512 + d] = (_Float16)y;
    __syncthreads();
  }
}
__global__ __launch_bounds__(256) void rmsnorm_kernel(
    const float* __restrict__ h, const float* __restrict__ w,
    _Float16* __restrict__ o, int C)
{
  int n = blockIdx.x, c = threadIdx.x;
  float v = h[(size_t)n * C + c];
  float ss = v * v;
#pragma unroll
  for (int off = 16; off; off >>= 1) ss += __shfl_xor(ss, off);
  __shared__ float red[8];
  if ((c & 31) == 0) red[c >> 5] = ss;
  __syncthreads();
  float t = 0.f;
#pragma unroll
  for (int i = 0; i < 8; ++i) t += red[i];
  o[(size_t)n * C + c] = (_Float16)(v * rsqrtf(t / C + 1e-5f) * w[c]);
}
__global__ void head2_kernel(const _Float16* __restrict__ e2,
                             const float* __restrict__ w2, const float* __restrict__ b2,
                             float* __restrict__ out, int N)
{
  int n = blockIdx.x * blockDim.x + threadIdx.x;
  if (n >= N) return;
  float a0 = b2[0], a1 = b2[1];
#pragma unroll 4
  for (int k = 0; k < 128; ++k) {
    float e = (float)e2[(size_t)n * 128 + k];
    a0 += e * w2[k];
    a1 += e * w2[128 + k];
  }
  out[(size_t)n * 2 + 0] = a0;
  out[(size_t)n * 2 + 1] = a1;
}

// ============================================================
// host launcher
// ============================================================
extern "C" void kernel_launch(void* const* d_in, const int* in_sizes, int n_in,
                              void* d_out, int out_size, void* d_ws, size_t ws_size,
                              hipStream_t stream) {
  (void)in_sizes; (void)n_in; (void)out_size; (void)ws_size;
  const int N = NTOK;

  size_t cur = 0;
  auto alloc = [&](size_t bytes) -> void* {
    void* p = (char*)d_ws + cur;
    cur += (bytes + 255) & ~(size_t)255;
    return p;
  };
  auto a16 = [&](size_t elems) { return (_Float16*)alloc(elems * 2); };
  auto a32 = [&](size_t elems) { return (float*)alloc(elems * 4); };

  _Float16 *w0a = a16(3ull*1024*1024), *w0b = a16(3ull*1024*1024);
  _Float16 *w1a = a16(3ull*768*1024),  *w1b = a16(3ull*768*768),  *w1d = a16(768ull*1024);
  _Float16 *w2a = a16(3ull*512*768),   *w2b = a16(3ull*512*512),  *w2d = a16(512ull*768);
  _Float16 *w3a = a16(3ull*256*512),   *w3b = a16(3ull*256*256),  *w3d = a16(256ull*512);
  _Float16 *inp[2], *xp[2], *dtp[2], *outp[2]; float* Af[2];
  for (int li = 0; li < 2; ++li) {
    inp[li]  = a16(1024ull*256);
    xp[li]   = a16(64ull*512);
    dtp[li]  = a16(512ull*32);
    outp[li] = a16(256ull*512);
    Af[li]   = a32(512*16);
  }
  _Float16 *outw = a16(256ull*256), *hw1 = a16(128ull*256);
  _Float16 *zpad = a16(64);
  _Float16 *act0 = a16((size_t)N*1024), *t1 = a16((size_t)N*1024), *t2 = a16((size_t)N*1024);
  _Float16 *act1 = a16((size_t)N*1024), *act2 = a16((size_t)N*768);
  _Float16 *act3 = a16((size_t)N*512),  *act4 = a16((size_t)N*256);
  float *hA = a32((size_t)N*256), *hB = a32((size_t)N*256);
  _Float16 *nrm  = a16((size_t)N*256),  *xz  = a16((size_t)N*1024);
  _Float16 *xc   = a16((size_t)N*512),  *xdbl = a16((size_t)N*64);
  _Float16 *dt16 = a16((size_t)N*512),  *y16 = a16((size_t)N*512);
  _Float16 *e1   = a16((size_t)N*256),  *e2  = a16((size_t)N*128);

  const float* X    = (const float*)d_in[0];
  const float *W01=(const float*)d_in[1],  *B01=(const float*)d_in[2],
              *W02=(const float*)d_in[3],  *B02=(const float*)d_in[4];
  const float *W11=(const float*)d_in[5],  *B11=(const float*)d_in[6],
              *W12=(const float*)d_in[7],  *B12=(const float*)d_in[8],
              *WD1=(const float*)d_in[9],  *BD1=(const float*)d_in[10];
  const float *W21=(const float*)d_in[11], *B21=(const float*)d_in[12],
              *W22=(const float*)d_in[13], *B22=(const float*)d_in[14],
              *WD2=(const float*)d_in[15], *BD2=(const float*)d_in[16];
  const float *W31=(const float*)d_in[17], *B31=(const float*)d_in[18],
              *W32=(const float*)d_in[19], *B32=(const float*)d_in[20],
              *WD3=(const float*)d_in[21], *BD3=(const float*)d_in[22];
  const float *norm_w[2], *in_proj[2], *conv_w[2], *conv_b[2], *x_proj[2],
              *dt_proj[2], *dt_b[2], *A_log[2], *Dparam[2], *out_proj[2];
  for (int li = 0; li < 2; ++li) {
    int b = 23 + li * 10;
    norm_w[li]=(const float*)d_in[b+0]; in_proj[li]=(const float*)d_in[b+1];
    conv_w[li]=(const float*)d_in[b+2]; conv_b[li]=(const float*)d_in[b+3];
    x_proj[li]=(const float*)d_in[b+4]; dt_proj[li]=(const float*)d_in[b+5];
    dt_b[li]=(const float*)d_in[b+6];   A_log[li]=(const float*)d_in[b+7];
    Dparam[li]=(const float*)d_in[b+8]; out_proj[li]=(const float*)d_in[b+9];
  }
  const float *NFW=(const float*)d_in[43], *OUTW=(const float*)d_in[44], *OUTB=(const float*)d_in[45];
  const float *HW1=(const float*)d_in[46], *HB1=(const float*)d_in[47],
              *HW2=(const float*)d_in[48], *HB2=(const float*)d_in[49];
  float* out = (float*)d_out;

  auto ew = [&](int n) { return dim3((n + 255) / 256); };

  zero_pad_kernel<<<1,64,0,stream>>>(zpad);
  convw_to_f16<<<ew(1024*1024*3),256,0,stream>>>(W01, w0a, 1024, 1024);
  convw_to_f16<<<ew(1024*1024*3),256,0,stream>>>(W02, w0b, 1024, 1024);
  convw_to_f16<<<ew(768*1024*3),256,0,stream>>>(W11, w1a, 768, 1024);
  convw_to_f16<<<ew(768*768*3),256,0,stream>>>(W12, w1b, 768, 768);
  mat_pad_f16<<<ew(768*1024),256,0,stream>>>(WD1, w1d, 768, 1024, 768, 1024);
  convw_to_f16<<<ew(512*768*3),256,0,stream>>>(W21, w2a, 512, 768);
  convw_to_f16<<<ew(512*512*3),256,0,stream>>>(W22, w2b, 512, 512);
  mat_pad_f16<<<ew(512*768),256,0,stream>>>(WD2, w2d, 512, 768, 512, 768);
  convw_to_f16<<<ew(256*512*3),256,0,stream>>>(W31, w3a, 256, 512);
  convw_to_f16<<<ew(256*256*3),256,0,stream>>>(W32, w3b, 256, 256);
  mat_pad_f16<<<ew(256*512),256,0,stream>>>(WD3, w3d, 256, 512, 256, 512);
  for (int li = 0; li < 2; ++li) {
    mat_pad_f16<<<ew(1024*256),256,0,stream>>>(in_proj[li], inp[li], 1024, 256, 1024, 256);
    mat_pad_f16<<<ew(64*512),256,0,stream>>>(x_proj[li], xp[li], 48, 512, 64, 512);
    mat_pad_f16<<<ew(512*32),256,0,stream>>>(dt_proj[li], dtp[li], 512, 16, 512, 32);
    mat_pad_f16<<<ew(256*512),256,0,stream>>>(out_proj[li], outp[li], 256, 512, 256, 512);
    a_prep<<<ew(512*16),256,0,stream>>>(A_log[li], Af[li], 512*16);
  }
  mat_pad_f16<<<ew(256*256),256,0,stream>>>(OUTW, outw, 256, 256, 256, 256);
  mat_pad_f16<<<ew(128*256),256,0,stream>>>(HW1, hw1, 128, 256, 128, 256);

  auto gemm = [&](const _Float16* W, const _Float16* Xa, const float* bias,
                  const _Float16* add16, const float* addf,
                  _Float16* o16, float* of,
                  int M, int K, int in_stride, int out_stride,
                  int taps, int dil, int flags) {
    dim3 g(N / 128, M / 64), blk(256);
    wmma_gemm_kernel<<<g, blk, 0, stream>>>(W, Xa, bias, add16, addf, zpad, o16, of,
                                            M, K, in_stride, out_stride, taps, dil, flags);
  };

  // ---- TCN ----
  cast_f32_to_f16<<<ew(N*1024),256,0,stream>>>(X, act0, N*1024);
  gemm(w0a, act0, B01, nullptr, nullptr, t1, nullptr, 1024, 1024, 1024, 1024, 3, 1, FLAG_RELU);
  gemm(w0b, t1,   B02, nullptr, nullptr, t2, nullptr, 1024, 1024, 1024, 1024, 3, 1, FLAG_RELU);
  add_relu_f16<<<ew(N*1024),256,0,stream>>>(t2, act0, act1, N*1024);
  gemm(w1a, act1, B11, nullptr, nullptr, t1, nullptr, 768, 1024, 1024, 768, 3, 2, FLAG_RELU);
  gemm(w1b, t1,   B12, nullptr, nullptr, t2, nullptr, 768, 768,  768,  768, 3, 2, FLAG_RELU);
  gemm(w1d, act1, BD1, t2, nullptr, act2, nullptr, 768, 1024, 1024, 768, 1, 0, FLAG_RELU);
  gemm(w2a, act2, B21, nullptr, nullptr, t1, nullptr, 512, 768, 768, 512, 3, 4, FLAG_RELU);
  gemm(w2b, t1,   B22, nullptr, nullptr, t2, nullptr, 512, 512, 512, 512, 3, 4, FLAG_RELU);
  gemm(w2d, act2, BD2, t2, nullptr, act3, nullptr, 512, 768, 768, 512, 1, 0, FLAG_RELU);
  gemm(w3a, act3, B31, nullptr, nullptr, t1, nullptr, 256, 512, 512, 256, 3, 8, FLAG_RELU);
  gemm(w3b, t1,   B32, nullptr, nullptr, t2, nullptr, 256, 256, 256, 256, 3, 8, FLAG_RELU);
  gemm(w3d, act3, BD3, t2, nullptr, act4, nullptr, 256, 512, 512, 256, 1, 0, FLAG_RELU);
  cast_f16_to_f32<<<ew(N*256),256,0,stream>>>(act4, hA, N*256);

  // ---- Mamba layers ----
  float* hin = hA;
  float* hout = hB;
  for (int li = 0; li < 2; ++li) {
    rmsnorm_kernel<<<N,256,0,stream>>>(hin, norm_w[li], nrm, 256);
    gemm(inp[li], nrm, nullptr, nullptr, nullptr, xz, nullptr, 1024, 256, 256, 1024, 1, 0, 0);
    dwconv_silu<<<ew(N*512),256,0,stream>>>(xz, conv_w[li], conv_b[li], xc, N);
    gemm(xp[li], xc, nullptr, nullptr, nullptr, xdbl, nullptr, 64, 512, 512, 64, 1, 0, 0);
    gemm(dtp[li], xdbl, dt_b[li], nullptr, nullptr, dt16, nullptr, 512, 32, 64, 512, 1, 0, FLAG_SOFTPLUS);
    scan_kernel<<<8,512,0,stream>>>(xdbl, dt16, xc, xz, Af[li], Dparam[li], y16);
    gemm(outp[li], y16, nullptr, nullptr, hin, nullptr, hout, 256, 512, 512, 256, 1, 0, 0);
    float* tswap = hin; hin = hout; hout = tswap;
  }

  // ---- head ----
  rmsnorm_kernel<<<N,256,0,stream>>>(hin, NFW, nrm, 256);
  gemm(outw, nrm, OUTB, nullptr, nullptr, e1, nullptr, 256, 256, 256, 256, 1, 0, 0);
  gemm(hw1,  e1,  HB1,  nullptr, nullptr, e2, nullptr, 128, 256, 256, 128, 1, 0, FLAG_RELU);
  head2_kernel<<<ew(N),256,0,stream>>>(e2, HW2, HB2, out, N);
}
